// NF_52432960750081
// MI455X (gfx1250) — compile-verified, matrix operand order checked by reading, not run on hardware
//
#include <hip/hip_runtime.h>

typedef __attribute__((ext_vector_type(16))) _Float16 v16h;
typedef __attribute__((ext_vector_type(8)))  _Float16 v8h;
typedef __attribute__((ext_vector_type(8)))  float    v8f;
typedef __attribute__((ext_vector_type(4)))  float    v4f;

#define NF_NLAYERS 8
#define NF_LATENT  64
#define NF_HALF    32
#define NF_HIDDEN  128
#define NF_SCALE   0.1f

#define NF_THREADS 512
#define NF_WAVES   (NF_THREADS / 32)
#define NF_ROWS_PER_BLOCK (NF_WAVES * 2 * 16)   // 2 tiles/wave -> 512 rows/block

// LDS leading dims (halves), rows padded by 8 halves (16B) to spread banks
#define LD1 40    // W1t: 128 rows x 32 k
#define LD2 136   // W2t: 128 rows x 128 k
#define LD3 136   // W3t:  64 rows x 128 k

#define W1_ELEMS (NF_HALF * NF_HIDDEN)     // 4096
#define W2_ELEMS (NF_HIDDEN * NF_HIDDEN)   // 16384
#define W3_ELEMS (NF_HIDDEN * NF_LATENT)   // 8192
#define NSUB     (NF_NLAYERS * 2)          // 16

// ---------- pre-pass: f32 weights -> transposed f16 in workspace ----------
// tW1[w][m][k] = W1[w][k][m], tW2[w][m][k] = W2[w][k][m], tW3[w][m][k] = W3[w][k][m]
__global__ void __launch_bounds__(256)
nf_convert_weights(const float* __restrict__ W1, const float* __restrict__ W2,
                   const float* __restrict__ W3,
                   _Float16* __restrict__ tW1, _Float16* __restrict__ tW2,
                   _Float16* __restrict__ tW3)
{
    const int i = blockIdx.x * 256 + threadIdx.x;
    const int n1 = NSUB * W1_ELEMS;            // 65536
    const int n2 = NSUB * W2_ELEMS;            // 262144
    const int n3 = NSUB * W3_ELEMS;            // 131072
    if (i < n1) {
        int w = i >> 12, r = i & 4095;
        int m = r >> 5, k = r & 31;            // r = m*32 + k
        tW1[i] = (_Float16)W1[w * W1_ELEMS + k * NF_HIDDEN + m];
    } else if (i < n1 + n2) {
        int j = i - n1;
        int w = j >> 14, r = j & 16383;
        int m = r >> 7, k = r & 127;           // r = m*128 + k
        tW2[j] = (_Float16)W2[w * W2_ELEMS + k * NF_HIDDEN + m];
    } else if (i < n1 + n2 + n3) {
        int j = i - n1 - n2;
        int w = j >> 13, r = j & 8191;
        int m = r >> 7, k = r & 127;           // r = m*128 + k, m<64
        tW3[j] = (_Float16)W3[w * W3_ELEMS + k * NF_LATENT + m];
    }
}

// ---------- fragment helpers ----------
// A fragment (16x32 f16, Wt rows): lane<16 -> m=lane, K {0..7,16..23}; lane>=16 -> K {8..15,24..31}
__device__ __forceinline__ v16h nf_ldsA(const _Float16* W, int ld, int mtile, int kchunk, int lane)
{
    const int row = mtile * 16 + (lane & 15);
    const int kb  = kchunk * 32 + ((lane & 16) ? 8 : 0);
    const _Float16* p = W + row * ld + kb;
    v8h lo = *(const v8h*)p;
    v8h hi = *(const v8h*)(p + 16);
    v16h a;
#pragma unroll
    for (int j = 0; j < 8; ++j) { a[j] = lo[j]; a[j + 8] = hi[j]; }
    return a;
}

// Bias as C fragment: lane group picks M half of the 16-tile
__device__ __forceinline__ v8f nf_ldsBias(const float* B, int mtile, int hi8)
{
    const float* p = B + mtile * 16 + hi8;
    v8f c;
#pragma unroll
    for (int j = 0; j < 8; ++j) c[j] = p[j];
    return c;
}

// Two f32 D fragments (M tiles 2c, 2c+1) -> one f16 B fragment (K = 32c..32c+31). Lane-local.
__device__ __forceinline__ v16h nf_packB(v8f lo, v8f hi)
{
    v16h b;
#pragma unroll
    for (int j = 0; j < 8; ++j) { b[j] = (_Float16)lo[j]; b[j + 8] = (_Float16)hi[j]; }
    return b;
}

// relu in packed f16 (v_pk_max_num_f16): 2 elems/op
__device__ __forceinline__ v16h nf_relu16(v16h b)
{
#if __has_builtin(__builtin_elementwise_max)
    const v16h z = {};
    return __builtin_elementwise_max(b, z);
#else
    v16h r;
#pragma unroll
    for (int j = 0; j < 16; ++j) r[j] = b[j] > (_Float16)0.0f ? b[j] : (_Float16)0.0f;
    return r;
#endif
}

__device__ __forceinline__ v16h nf_packB_relu(v8f lo, v8f hi)
{
    return nf_relu16(nf_packB(lo, hi));
}

__device__ __forceinline__ float nf_tanh(float v)
{
#if __has_builtin(__builtin_amdgcn_tanhf)
    return __builtin_amdgcn_tanhf(v);
#else
    return tanhf(v);
#endif
}

#define NF_WMMA(a, b, c) \
    __builtin_amdgcn_wmma_f32_16x16x32_f16(false, (a), false, (b), (short)0, (c), false, false)

// register-resident staging buffer: 7 x b128 of weights + 1 bias float per thread
struct NFStage {
    v8h r1;      // 1 chunk of W1t  (4096  halves / 512 threads)
    v8h r2[4];   // 4 chunks of W2t (16384 halves / 512 threads)
    v8h r3[2];   // 2 chunks of W3t (8192  halves / 512 threads)
    float rb;    // one bias element (tid < 320)
};

__device__ __forceinline__ void nf_stage_load(NFStage& st, int s, int tid,
                                              const _Float16* tW1, const _Float16* tW2,
                                              const _Float16* tW3,
                                              const float* b1, const float* b2, const float* b3)
{
    const _Float16* g1 = tW1 + (long)s * W1_ELEMS;
    const _Float16* g2 = tW2 + (long)s * W2_ELEMS;
    const _Float16* g3 = tW3 + (long)s * W3_ELEMS;
    st.r1 = *(const v8h*)&g1[tid * 8];
#pragma unroll
    for (int q = 0; q < 4; ++q) st.r2[q] = *(const v8h*)&g2[(tid + q * NF_THREADS) * 8];
#pragma unroll
    for (int q = 0; q < 2; ++q) st.r3[q] = *(const v8h*)&g3[(tid + q * NF_THREADS) * 8];
    float rb = 0.0f;
    if (tid < 128)      rb = b1[(long)s * NF_HIDDEN + tid];
    else if (tid < 256) rb = b2[(long)s * NF_HIDDEN + (tid - 128)];
    else if (tid < 320) rb = b3[(long)s * NF_LATENT + (tid - 256)];
    st.rb = rb;
}

// ---------- main fused flow kernel ----------
__global__ void __launch_bounds__(NF_THREADS)
nf_flow_kernel(const float* __restrict__ x,
               const _Float16* __restrict__ tW1, const float* __restrict__ b1,
               const _Float16* __restrict__ tW2, const float* __restrict__ b2,
               const _Float16* __restrict__ tW3, const float* __restrict__ b3,
               float* __restrict__ out)
{
    __shared__ __align__(16) _Float16 sW1[NF_HIDDEN * LD1]; // 10240 B
    __shared__ __align__(16) _Float16 sW2[NF_HIDDEN * LD2]; // 34816 B
    __shared__ __align__(16) _Float16 sW3[NF_LATENT * LD3]; // 17408 B
    __shared__ __align__(16) float sB1[NF_HIDDEN];
    __shared__ __align__(16) float sB2[NF_HIDDEN];
    __shared__ __align__(16) float sB3[NF_LATENT];

    const int tid  = threadIdx.x;
    const int lane = tid & 31;
    const int wave = tid >> 5;
    const int hi8  = (lane & 16) ? 8 : 0;

    // each wave owns two 16-row tiles; batch row = lane&15 within a tile
    const long row0 = (long)blockIdx.x * NF_ROWS_PER_BLOCK + (wave * 2 + 0) * 16 + (lane & 15);
    const long row1 = (long)blockIdx.x * NF_ROWS_PER_BLOCK + (wave * 2 + 1) * 16 + (lane & 15);

    // activations in D-fragment layout: frag f covers dims f*16 + hi8 + 0..7
    // frags 0..1 = "input half", frags 2..3 = "other half" (swapped each sub-step)
    v8f xA[4], xB[4];
    {
        const float* xr0 = x + row0 * NF_LATENT;
        const float* xr1 = x + row1 * NF_LATENT;
#pragma unroll
        for (int f = 0; f < 4; ++f) {
            const int d0 = f * 16 + hi8;
            v4f a0 = *(const v4f*)(xr0 + d0);
            v4f b0 = *(const v4f*)(xr0 + d0 + 4);
            v4f a1 = *(const v4f*)(xr1 + d0);
            v4f b1v = *(const v4f*)(xr1 + d0 + 4);
#pragma unroll
            for (int j = 0; j < 4; ++j) {
                xA[f][j] = a0[j]; xA[f][j + 4] = b0[j];
                xB[f][j] = a1[j]; xB[f][j + 4] = b1v[j];
            }
        }
    }

    // prefetch sub-step 0 weights into registers
    NFStage st;
    nf_stage_load(st, 0, tid, tW1, tW2, tW3, b1, b2, b3);

    for (int s = 0; s < NSUB; ++s) {
        __syncthreads();   // previous sub-step done reading LDS
        // ---- commit prefetched registers to padded LDS (ds_store only) ----
        {
            const int i1 = tid * 8;
            *(v8h*)&sW1[(i1 >> 5) * LD1 + (i1 & 31)] = st.r1;
#pragma unroll
            for (int q = 0; q < 4; ++q) {
                const int i = (tid + q * NF_THREADS) * 8;
                *(v8h*)&sW2[(i >> 7) * LD2 + (i & 127)] = st.r2[q];
            }
#pragma unroll
            for (int q = 0; q < 2; ++q) {
                const int i = (tid + q * NF_THREADS) * 8;
                *(v8h*)&sW3[(i >> 7) * LD3 + (i & 127)] = st.r3[q];
            }
            if (tid < 128)      sB1[tid] = st.rb;
            else if (tid < 256) sB2[tid - 128] = st.rb;
            else if (tid < 320) sB3[tid - 256] = st.rb;
        }
        // issue next sub-step's global loads now; they complete during compute below
        if (s + 1 < NSUB)
            nf_stage_load(st, s + 1, tid, tW1, tW2, tW3, b1, b2, b3);
        __syncthreads();   // staging visible to all waves

        // input half -> f16 B fragments (16 rows x 32 dims), lane-local pack
        v16h binA = nf_packB(xA[0], xA[1]);
        v16h binB = nf_packB(xB[0], xB[1]);

        // ---- layer 1: 32 -> 128; M-tile pairs, pack+relu immediately ----
        v16h p1A[4], p1B[4];
#pragma unroll
        for (int g = 0; g < 4; ++g) {
            v8f cb0 = nf_ldsBias(sB1, 2 * g, hi8);
            v8f cb1 = nf_ldsBias(sB1, 2 * g + 1, hi8);
            v16h a0 = nf_ldsA(sW1, LD1, 2 * g, 0, lane);
            v16h a1 = nf_ldsA(sW1, LD1, 2 * g + 1, 0, lane);
            v8f hA0 = NF_WMMA(a0, binA, cb0);
            v8f hB0 = NF_WMMA(a0, binB, cb0);
            v8f hA1 = NF_WMMA(a1, binA, cb1);
            v8f hB1 = NF_WMMA(a1, binB, cb1);
            p1A[g] = nf_packB_relu(hA0, hA1);
            p1B[g] = nf_packB_relu(hB0, hB1);
        }

        // ---- layer 2: 128 -> 128; M-tile pairs x 4 K chunks (16 WMMAs/group) ----
        v16h p2A[4], p2B[4];
#pragma unroll
        for (int g = 0; g < 4; ++g) {
            v8f c0 = nf_ldsBias(sB2, 2 * g, hi8);
            v8f c1 = c0;
            v8f d0 = nf_ldsBias(sB2, 2 * g + 1, hi8);
            v8f d1 = d0;
#pragma unroll
            for (int k = 0; k < 4; ++k) {
                v16h a0 = nf_ldsA(sW2, LD2, 2 * g, k, lane);
                c0 = NF_WMMA(a0, p1A[k], c0);
                c1 = NF_WMMA(a0, p1B[k], c1);
                v16h a1 = nf_ldsA(sW2, LD2, 2 * g + 1, k, lane);
                d0 = NF_WMMA(a1, p1A[k], d0);
                d1 = NF_WMMA(a1, p1B[k], d1);
            }
            p2A[g] = nf_packB_relu(c0, d0);
            p2B[g] = nf_packB_relu(c1, d1);
        }

        // ---- layer 3: 128 -> 64 (4 M tiles x 4 K chunks) ----
        v8f stA[4], stB[4];
#pragma unroll
        for (int t = 0; t < 4; ++t) {
            v8f c0 = nf_ldsBias(sB3, t, hi8);
            v8f c1 = c0;
#pragma unroll
            for (int k = 0; k < 4; ++k) {
                v16h a = nf_ldsA(sW3, LD3, t, k, lane);
                c0 = NF_WMMA(a, p2A[k], c0);
                c1 = NF_WMMA(a, p2B[k], c1);
            }
            stA[t] = c0;
            stB[t] = c1;
        }

        // ---- coupling: other = other * exp(SCALE*tanh(s)) + t ; then swap halves ----
#pragma unroll
        for (int f = 0; f < 2; ++f) {
#pragma unroll
            for (int j = 0; j < 8; ++j) {
                float eA = __expf(NF_SCALE * nf_tanh(stA[f][j]));
                float eB = __expf(NF_SCALE * nf_tanh(stB[f][j]));
                xA[2 + f][j] = xA[2 + f][j] * eA + stA[2 + f][j];
                xB[2 + f][j] = xB[2 + f][j] * eB + stB[2 + f][j];
            }
        }
#pragma unroll
        for (int f = 0; f < 2; ++f) {
            v8f tA = xA[f]; xA[f] = xA[2 + f]; xA[2 + f] = tA;
            v8f tB = xB[f]; xB[f] = xB[2 + f]; xB[2 + f] = tB;
        }
    }

    // after an even number of sub-steps the halves are back in (y1, y2) order
    {
        float* o0 = out + row0 * NF_LATENT;
        float* o1 = out + row1 * NF_LATENT;
#pragma unroll
        for (int f = 0; f < 4; ++f) {
            const int d0 = f * 16 + hi8;
            v4f a0, b0, a1, b1v;
#pragma unroll
            for (int j = 0; j < 4; ++j) {
                a0[j] = xA[f][j]; b0[j] = xA[f][j + 4];
                a1[j] = xB[f][j]; b1v[j] = xB[f][j + 4];
            }
            *(v4f*)(o0 + d0)     = a0;
            *(v4f*)(o0 + d0 + 4) = b0;
            *(v4f*)(o1 + d0)     = a1;
            *(v4f*)(o1 + d0 + 4) = b1v;
        }
    }
}

extern "C" void kernel_launch(void* const* d_in, const int* in_sizes, int n_in,
                              void* d_out, int out_size, void* d_ws, size_t ws_size,
                              hipStream_t stream)
{
    const float* x  = (const float*)d_in[0];
    const float* W1 = (const float*)d_in[1];
    const float* b1 = (const float*)d_in[2];
    const float* W2 = (const float*)d_in[3];
    const float* b2 = (const float*)d_in[4];
    const float* W3 = (const float*)d_in[5];
    const float* b3 = (const float*)d_in[6];
    float* out = (float*)d_out;

    // workspace: transposed f16 weights (16 sub-steps), 917504 bytes total
    _Float16* tW1 = (_Float16*)d_ws;
    _Float16* tW2 = tW1 + NSUB * W1_ELEMS;
    _Float16* tW3 = tW2 + NSUB * W2_ELEMS;

    const int total_w = NSUB * (W1_ELEMS + W2_ELEMS + W3_ELEMS); // 458752
    nf_convert_weights<<<(total_w + 255) / 256, 256, 0, stream>>>(W1, W2, W3, tW1, tW2, tW3);

    const int batch  = in_sizes[0] / NF_LATENT;            // 262144
    const int blocks = batch / NF_ROWS_PER_BLOCK;          // 512
    nf_flow_kernel<<<blocks, NF_THREADS, 0, stream>>>(x, tW1, b1, tW2, b2, tW3, b3, out);
}
